// PhaseDetNet_60112362275539
// MI455X (gfx1250) — matrix operand order repaired and checked by reading, not compiled, not run on hardware
//
#include <hip/hip_runtime.h>
#include <hip/hip_bf16.h>

typedef __attribute__((ext_vector_type(16))) _Float16 v16h;
typedef __attribute__((ext_vector_type(8)))  float    v8f;

#define BATCH 16
#define BLK   1024
#define S     1032
#define TWO_S 2064
#define OH    4
#define V_LEN 128
#define Z_LEN 256
#define KDIM  (S + V_LEN)     // 1160
#define OHN   (S * OH)        // 4128
#define NCH   4               // j-chunks in pass 2 (2048/4 = 512 rows each)

// ---------------- workspace layout (float slots) ----------------
#define OFF_X    0                         // 16*2064        = 33024
#define OFF_C    33024                     // 16*1032        = 16512
#define OFF_SN   49536                     // 16*1032        = 16512
#define OFF_PX   66048                     // 2*16*2048      = 65536
#define OFF_W    131584                    // 2*16*2048      = 65536
#define OFF_TP   197120                    // NCH*2*16*2064  = 264192
#define OFF_Q    461312                    // 16*1032        = 16512
#define OFF_ZH   477824                    // 16*256 halves  = 2048 float slots
#define OFF_W2H  479872                    // 4128*256 halves= 528384 float slots
#define OFF_W3H  1008256                   // 128*256 halves = 16384 float slots

// ---------------- prepare: c, s, x = [mag*c, mag*s] ----------------
__global__ void k_prep(const float* __restrict__ x_phase, const float* __restrict__ x_mag,
                       float* __restrict__ C, float* __restrict__ Sn, float* __restrict__ X) {
    int idx = blockIdx.x * blockDim.x + threadIdx.x;
    if (idx >= BATCH * S) return;
    int b = idx / S, i = idx - b * S;
    float p = x_phase[idx], mg = x_mag[idx];
    float cc = cosf(p), ss = sinf(p);
    C[idx] = cc; Sn[idx] = ss;
    X[b * TWO_S + i]     = mg * cc;
    X[b * TWO_S + S + i] = mg * ss;
}

// ---------------- f32 -> f16 conversion ----------------
__global__ void k_cvt(const float* __restrict__ src, _Float16* __restrict__ dst, int n) {
    int i = blockIdx.x * blockDim.x + threadIdx.x;
    if (i < n) dst[i] = (_Float16)src[i];
}

// ---------------- pass 1: Px[mat,b,m] = Psi[b,m,:] . x[b,:] ----------------
__global__ void k_rowdot(const float* __restrict__ Psi_e, const float* __restrict__ Psi_o,
                         const float* __restrict__ X, float* __restrict__ PX) {
    __shared__ float red[256];
    int bi  = blockIdx.x;            // 0 .. 2*16*2048-1
    int mat = bi >> 15;
    int rem = bi & 32767;
    int b    = rem >> 11;
    int mrow = rem & 2047;
    const float* Psi = mat ? Psi_o : Psi_e;
    const float4* row = (const float4*)(Psi + ((size_t)(b * 2048 + mrow)) * TWO_S);
    const float4* xv  = (const float4*)(X + b * TWO_S);
    int t = threadIdx.x;
    float acc = 0.f;
    for (int i = t; i < TWO_S / 4; i += 256) {       // 516 float4 per row
        float4 a = row[i], c4 = xv[i];
        acc += a.x * c4.x + a.y * c4.y + a.z * c4.z + a.w * c4.w;
    }
    red[t] = acc; __syncthreads();
    for (int sft = 128; sft > 0; sft >>= 1) {
        if (t < sft) red[t] += red[t + sft];
        __syncthreads();
    }
    if (t == 0) PX[bi] = red[0];
}

// ---------------- per-row weights: w = u * (rhs_coef*y + sql_coef*(u1^2+u2^2)) ----------------
__global__ void k_weights(const float* __restrict__ PX,
                          const float* __restrict__ y_e, const float* __restrict__ y_o,
                          const float* __restrict__ d1, const float* __restrict__ d2,
                          const float* __restrict__ d4, float* __restrict__ Wv) {
    int idx = blockIdx.x * blockDim.x + threadIdx.x;  // 2*16*1024
    if (idx >= 2 * BATCH * BLK) return;
    int mat = idx >> 14;
    int rem = idx & 16383;
    int b = rem >> 10, j = rem & 1023;
    int base = mat * 32768 + b * 2048;
    float u1 = PX[base + j];
    float u2 = PX[base + 1024 + j];
    float yy = (mat ? y_o : y_e)[b * BLK + j];
    float rc = mat ? -d2[0] : -d1[0];
    float sc = mat ?  d4[0] :  d2[0];
    float r  = rc * yy + sc * (u1 * u1 + u2 * u2);
    Wv[base + j]        = u1 * r;
    Wv[base + 1024 + j] = u2 * r;
}

// ---------------- pass 2: partial t[b,m] = sum_j Psi[b,j,m]*w[j] over a 512-row chunk ----------------
__global__ void k_coldot(const float* __restrict__ Psi_e, const float* __restrict__ Psi_o,
                         const float* __restrict__ Wv, float* __restrict__ TP) {
    __shared__ float wl[512];
    int bi = blockIdx.x;             // 2*16*9*NCH = 1152
    int ch = bi & 3;  bi >>= 2;
    int tile = bi % 9; bi /= 9;
    int b = bi & 15;
    int mat = bi >> 4;
    const float* Psi = mat ? Psi_o : Psi_e;
    int m = tile * 256 + threadIdx.x;
    for (int i = threadIdx.x; i < 512; i += 256)
        wl[i] = Wv[mat * 32768 + b * 2048 + ch * 512 + i];
    __syncthreads();
    float acc = 0.f;
    if (m < TWO_S) {
        const float* base = Psi + ((size_t)(b * 2048 + ch * 512)) * TWO_S + m;
        for (int jj = 0; jj < 512; ++jj) {
            __builtin_prefetch(base + (size_t)(jj + 8) * TWO_S, 0, 1);  // global_prefetch_b8
            acc += base[(size_t)jj * TWO_S] * wl[jj];
        }
        TP[(size_t)((ch * 2 + mat) * 16 + b) * TWO_S + m] = acc;
    }
}

// ---------------- q = x_phase + sum_{mat,ch} (-s*mag*t[:S] + c*mag*t[S:]) ----------------
__global__ void k_q(const float* __restrict__ x_phase, const float* __restrict__ x_mag,
                    const float* __restrict__ C, const float* __restrict__ Sn,
                    const float* __restrict__ TP, float* __restrict__ Q) {
    int idx = blockIdx.x * blockDim.x + threadIdx.x;
    if (idx >= BATCH * S) return;
    int b = idx / S, i = idx - b * S;
    float sm = Sn[idx] * x_mag[idx];
    float cm = C[idx]  * x_mag[idx];
    float acc = x_phase[idx];
    #pragma unroll
    for (int ch = 0; ch < NCH; ++ch)
        #pragma unroll
        for (int mat = 0; mat < 2; ++mat) {
            const float* tp = TP + (size_t)((ch * 2 + mat) * 16 + b) * TWO_S;
            acc += -sm * tp[i] + cm * tp[i + S];
        }
    Q[idx] = acc;
}

// ---------------- MLP layer 1 (scalar f32, trivial): z = relu([q,v] @ W1^T + b1) -> f16 ----------------
__global__ void k_mlp1(const float* __restrict__ Q, const float* __restrict__ V,
                       const float* __restrict__ W1, const float* __restrict__ b1,
                       _Float16* __restrict__ zh) {
    int idx = blockIdx.x * blockDim.x + threadIdx.x;   // 16*256
    if (idx >= BATCH * Z_LEN) return;
    int m = idx >> 8, n = idx & 255;
    const float* w = W1 + (size_t)n * KDIM;
    float acc = b1[n];
    for (int k = 0; k < S; ++k)     acc += Q[m * S + k] * w[k];
    for (int k = 0; k < V_LEN; ++k) acc += V[m * V_LEN + k] * w[S + k];
    acc = acc > 0.f ? acc : 0.f;
    zh[m * Z_LEN + n] = (_Float16)acc;
}

// ---------------- WMMA GEMM: out(16xN) = addm + A(16x256,f16) @ B(Nx256,f16)^T + bias ----------------
// One wave per 16x16 tile; K=256 in 8 steps of v_wmma_f32_16x16x32_f16.
__global__ void k_wmma_gemm(const _Float16* __restrict__ A, const _Float16* __restrict__ B,
                            const float* __restrict__ bias, const float* __restrict__ addm,
                            float* __restrict__ out, int N) {
    int wave = (blockIdx.x * blockDim.x + threadIdx.x) >> 5;
    int lane = threadIdx.x & 31;
    int ntile = wave * 16;
    if (ntile >= N) return;                 // whole-wave guard: EXEC stays all-ones
    int mloc = lane & 15;
    int half = lane >> 4;
    int ncol = ntile + mloc;
    v8f acc = {};
    for (int kk = 0; kk < Z_LEN; kk += 32) {
        v16h a, bfrag;
        #pragma unroll
        for (int v = 0; v < 8; ++v) {
            // A 16x32 (MxK) layout: lane m=lane&15; VGPR v holds K = (v&4?16:0)+half*8+(v&3)*2, +1
            int ka = kk + ((v & 4) ? 16 : 0) + half * 8 + (v & 3) * 2;
            a[2 * v]     = A[mloc * Z_LEN + ka];
            a[2 * v + 1] = A[mloc * Z_LEN + ka + 1];
            // B 32x16 (KxN) layout: lane n=lane&15; VGPR v holds K = (v&4?16:0)+half*8+(v&3)*2, +1
            int kb = kk + ((v & 4) ? 16 : 0) + half * 8 + (v & 3) * 2;
            bfrag[2 * v]     = B[(size_t)ncol * Z_LEN + kb];
            bfrag[2 * v + 1] = B[(size_t)ncol * Z_LEN + kb + 1];
        }
        acc = __builtin_amdgcn_wmma_f32_16x16x32_f16(
            /*neg_a=*/false, a, /*neg_b=*/false, bfrag,
            /*c_mod=*/(short)0, acc, /*reuse_a=*/false, /*reuse_b=*/false);
    }
    #pragma unroll
    for (int r = 0; r < 8; ++r) {           // C/D layout: VGPR r -> M = r + 8*half, N = lane&15
        int mrow = r + 8 * half;
        out[(size_t)mrow * N + ncol] = addm[(size_t)mrow * N + ncol] + acc[r] + bias[ncol];
    }
}

// ---------------- softmax over OH=4 + mapp projection ----------------
__global__ void k_soft(const float* __restrict__ oh, const float* __restrict__ mapp,
                       float* __restrict__ xpn) {
    int idx = blockIdx.x * blockDim.x + threadIdx.x;   // 16*1032
    if (idx >= BATCH * S) return;
    const float* p = oh + (size_t)idx * OH;
    float a0 = p[0], a1 = p[1], a2 = p[2], a3 = p[3];
    float mx = fmaxf(fmaxf(a0, a1), fmaxf(a2, a3));
    float e0 = expf(a0 - mx), e1 = expf(a1 - mx), e2 = expf(a2 - mx), e3 = expf(a3 - mx);
    float sum = e0 + e1 + e2 + e3;
    xpn[idx] = (e0 * mapp[0] + e1 * mapp[1] + e2 * mapp[2] + e3 * mapp[3]) / sum;
}

extern "C" void kernel_launch(void* const* d_in, const int* in_sizes, int n_in,
                              void* d_out, int out_size, void* d_ws, size_t ws_size,
                              hipStream_t stream) {
    const float* x_phase    = (const float*)d_in[1];
    const float* x_phase_oh = (const float*)d_in[2];
    const float* v          = (const float*)d_in[3];
    const float* x_mag      = (const float*)d_in[4];
    const float* y_e        = (const float*)d_in[5];
    const float* y_o        = (const float*)d_in[6];
    const float* Psi_e      = (const float*)d_in[7];
    const float* Psi_o      = (const float*)d_in[8];
    const float* mapp       = (const float*)d_in[9];
    const float* W1         = (const float*)d_in[10];
    const float* b1         = (const float*)d_in[11];
    const float* W2         = (const float*)d_in[12];
    const float* b2         = (const float*)d_in[13];
    const float* W3         = (const float*)d_in[14];
    const float* b3         = (const float*)d_in[15];
    const float* d1         = (const float*)d_in[16];
    const float* d2         = (const float*)d_in[17];
    const float* d4         = (const float*)d_in[19];

    float* ws = (float*)d_ws;
    float*     X   = ws + OFF_X;
    float*     C   = ws + OFF_C;
    float*     Sn  = ws + OFF_SN;
    float*     PX  = ws + OFF_PX;
    float*     Wv  = ws + OFF_W;
    float*     TP  = ws + OFF_TP;
    float*     Q   = ws + OFF_Q;
    _Float16*  zh  = (_Float16*)(ws + OFF_ZH);
    _Float16*  w2h = (_Float16*)(ws + OFF_W2H);
    _Float16*  w3h = (_Float16*)(ws + OFF_W3H);

    float* out     = (float*)d_out;
    float* out_xp  = out;                       // (16,1032)
    float* out_oh  = out + BATCH * S;           // (16,4128)
    float* out_v   = out + BATCH * S + BATCH * OHN;  // (16,128)

    // weight conversions (independent of data path)
    k_cvt<<<(OHN * Z_LEN + 255) / 256, 256, 0, stream>>>(W2, w2h, OHN * Z_LEN);
    k_cvt<<<(V_LEN * Z_LEN + 255) / 256, 256, 0, stream>>>(W3, w3h, V_LEN * Z_LEN);

    // phase prep
    k_prep<<<(BATCH * S + 255) / 256, 256, 0, stream>>>(x_phase, x_mag, C, Sn, X);

    // pass 1 over Psi: 2 * 16 * 2048 row dot-products
    k_rowdot<<<2 * BATCH * 2048, 256, 0, stream>>>(Psi_e, Psi_o, X, PX);

    // per-row weights
    k_weights<<<(2 * BATCH * BLK + 255) / 256, 256, 0, stream>>>(PX, y_e, y_o, d1, d2, d4, Wv);

    // pass 2 over Psi: chunked column reductions into deterministic partials
    k_coldot<<<2 * BATCH * 9 * NCH, 256, 0, stream>>>(Psi_e, Psi_o, Wv, TP);

    // q = x_phase + terms (reduces partials in fixed order)
    k_q<<<(BATCH * S + 255) / 256, 256, 0, stream>>>(x_phase, x_mag, C, Sn, TP, Q);

    // MLP layer 1 -> z (f16)
    k_mlp1<<<(BATCH * Z_LEN + 255) / 256, 256, 0, stream>>>(Q, v, W1, b1, zh);

    // x_phase_oh_new = x_phase_oh + z @ W2^T + b2   (WMMA, 258 tiles)
    {
        int tiles = OHN / 16;                  // 258
        int blocks = (tiles + 3) / 4;          // 4 waves / 128-thread block
        k_wmma_gemm<<<blocks, 128, 0, stream>>>(zh, w2h, b2, x_phase_oh, out_oh, OHN);
    }
    // v_new = v + z @ W3^T + b3   (WMMA, 8 tiles)
    k_wmma_gemm<<<2, 128, 0, stream>>>(zh, w3h, b3, v, out_v, V_LEN);

    // softmax over OH + mapp projection
    k_soft<<<(BATCH * S + 255) / 256, 256, 0, stream>>>(out_oh, mapp, out_xp);
}
